// RWKV_53051436040847
// MI455X (gfx1250) — matrix operand order repaired and checked by reading, not compiled, not run on hardware
//
#include <hip/hip_runtime.h>

// ---------------------------------------------------------------------------
// RWKV-v4 block forward for MI455X (gfx1250).
// bf16 WMMA GEMMs, 3-deep async global->LDS pipeline (ASYNCcnt, one barrier
// per stage), f32 WKV scan, fused epilogues.
// ---------------------------------------------------------------------------

typedef unsigned short u16;
typedef unsigned int   u32;

typedef __attribute__((ext_vector_type(16))) __bf16 v16bf;
typedef __attribute__((ext_vector_type(8)))  float  v8f;
typedef __attribute__((ext_vector_type(8)))  u32    v8u;

static constexpr int B_    = 8;
static constexpr int L_    = 2048;
static constexpr int SIZE_ = 1024;
static constexpr int ATT_  = 1024;
static constexpr int LIN_  = 4096;
static constexpr int NTOK  = B_ * L_;      // 16384 tokens

__device__ __forceinline__ u16 to_bf16(float f) {
    u32 u = __builtin_bit_cast(u32, f);
    u32 r = u + 0x7FFFu + ((u >> 16) & 1u);   // round-to-nearest-even
    return (u16)(r >> 16);
}

// --- CDNA5 async global->LDS copy (tracked by ASYNCcnt) --------------------
__device__ __forceinline__ void async_copy_b128(u32 lds_off, const void* gaddr) {
    asm volatile("global_load_async_to_lds_b128 %0, %1, off"
                 :: "v"(lds_off), "v"(gaddr) : "memory");
}
#define WAIT_ASYNCCNT(n) asm volatile("s_wait_asynccnt %0" :: "i"(n) : "memory")

// LDS byte offset of a __shared__ object (flat LDS aperture: addr[31:0]).
__device__ __forceinline__ u32 lds_offset(const void* p) {
    return (u32)(size_t)p;
}

// ---------------------------------------------------------------------------
// Weight convert+transpose: W[K][N] f32 -> Wt[N][K] bf16 (K-contiguous rows)
// ---------------------------------------------------------------------------
__global__ void transpose_bf16_kernel(const float* __restrict__ W,
                                      u16* __restrict__ Wt, int K, int N) {
    int idx = blockIdx.x * blockDim.x + threadIdx.x;
    if (idx >= K * N) return;
    int k = idx / N, n = idx - k * N;
    Wt[(size_t)n * K + k] = to_bf16(W[idx]);
}

// ---------------------------------------------------------------------------
// LayerNorm over last dim (1024). One block (256 thr) per token, 4 elems/thr.
// ---------------------------------------------------------------------------
__global__ __launch_bounds__(256) void ln_kernel(const float* __restrict__ x,
                                                 const float* __restrict__ g,
                                                 const float* __restrict__ b,
                                                 float* __restrict__ h) {
    __shared__ float s1[256], s2[256];
    const int tid = threadIdx.x;
    const size_t base = (size_t)blockIdx.x * SIZE_;
    float4 v = reinterpret_cast<const float4*>(x + base)[tid];
    float s = v.x + v.y + v.z + v.w;
    float q = v.x * v.x + v.y * v.y + v.z * v.z + v.w * v.w;
    s1[tid] = s; s2[tid] = q;
    __syncthreads();
    for (int off = 128; off > 0; off >>= 1) {
        if (tid < off) { s1[tid] += s1[tid + off]; s2[tid] += s2[tid + off]; }
        __syncthreads();
    }
    const float inv = 1.0f / (float)SIZE_;
    float mean = s1[0] * inv;
    float var  = s2[0] * inv - mean * mean;
    float rstd = rsqrtf(var + 1e-5f);
    float4 gg = reinterpret_cast<const float4*>(g)[tid];
    float4 bb = reinterpret_cast<const float4*>(b)[tid];
    float4 o;
    o.x = (v.x - mean) * rstd * gg.x + bb.x;
    o.y = (v.y - mean) * rstd * gg.y + bb.y;
    o.z = (v.z - mean) * rstd * gg.z + bb.z;
    o.w = (v.w - mean) * rstd * gg.w + bb.w;
    reinterpret_cast<float4*>(h + base)[tid] = o;
}

// ---------------------------------------------------------------------------
// Token-shift mixing: x* = h[t]*mix + h[t-1]*(1-mix), cast to bf16.
// ---------------------------------------------------------------------------
template <bool THREE>
__global__ __launch_bounds__(256) void mix_kernel(const float* __restrict__ h,
                                                  const float* __restrict__ mk,
                                                  const float* __restrict__ mv,
                                                  const float* __restrict__ mr,
                                                  u16* __restrict__ xk,
                                                  u16* __restrict__ xv,
                                                  u16* __restrict__ xr) {
    int idx = blockIdx.x * blockDim.x + threadIdx.x;          // < NTOK*SIZE
    int token = idx >> 10;
    int d     = idx & (SIZE_ - 1);
    float hv = h[idx];
    float hp = (token & (L_ - 1)) ? h[idx - SIZE_] : 0.0f;    // time-shift
    float a  = mk[d];
    xk[idx] = to_bf16(hv * a + hp * (1.0f - a));
    if constexpr (THREE) {
        float c = mv[d];
        xv[idx] = to_bf16(hv * c + hp * (1.0f - c));
    }
    float e = mr[d];
    xr[idx] = to_bf16(hv * e + hp * (1.0f - e));
}

// ---------------------------------------------------------------------------
// WKV recurrence: one lane per (b, d) channel; sequential over L.
// Fused: out = sigmoid(r_pre) * wkv, stored bf16 for the Wo GEMM.
// ---------------------------------------------------------------------------
__global__ __launch_bounds__(256) void wkv_kernel(const float* __restrict__ kb,
                                                  const float* __restrict__ vb,
                                                  const float* __restrict__ rb,
                                                  const float* __restrict__ td,
                                                  const float* __restrict__ tf,
                                                  u16* __restrict__ outb) {
    int idx = blockIdx.x * blockDim.x + threadIdx.x;          // < B_*ATT_
    int b = idx >> 10;
    int d = idx & (ATT_ - 1);
    float w = -expf(td[d]);
    float u = tf[d];
    float p = 0.0f, q = 0.0f, o = -1e38f;
    size_t off = (size_t)b * L_ * ATT_ + d;
    for (int t = 0; t < L_; ++t) {
        float kt = kb[off];
        float vt = vb[off];
        float no = fmaxf(o, u + kt);
        float A  = expf(o - no);
        float Bc = expf(u + kt - no);
        float y  = (A * p + Bc * vt) / (A * q + Bc);
        float r  = 1.0f / (1.0f + expf(-rb[off]));
        outb[off] = to_bf16(r * y);
        float no2 = fmaxf(o + w, kt);
        float A2  = expf(o + w - no2);
        float B2  = expf(kt - no2);
        p = A2 * p + B2 * vt;
        q = A2 * q + B2;
        o = no2;
        off += ATT_;
    }
}

// ---------------------------------------------------------------------------
// bf16 WMMA GEMM: C[M,N] = A[M,K] @ W[K,N], B supplied transposed Bt[N][K].
// Block: 256 thr (8 waves), tile 128x64, K-step 64, 3-deep LDS ring with
// async staging; exactly one workgroup barrier per K-stage.
// ---------------------------------------------------------------------------
enum { EPI_F32 = 0, EPI_ADD = 1, EPI_RELU2B = 2, EPI_SIG = 3, EPI_FINAL = 4 };

static constexpr int BK  = 64;   // K per stage (halves)
static constexpr int LDK = 72;   // padded LDS row stride (halves), 144B rows

template <int EPI>
__global__ __launch_bounds__(256) void gemm_bf16(const u16* __restrict__ A,
                                                 const u16* __restrict__ Bt,
                                                 float* __restrict__ Cf,
                                                 u16* __restrict__ Cb,
                                                 const float* __restrict__ Res,
                                                 const float* __restrict__ Scl,
                                                 int M, int N, int K) {
    __shared__ alignas(16) u16 As[3][128 * LDK];
    __shared__ alignas(16) u16 Bs[3][64 * LDK];

    const int tid  = threadIdx.x;
    const int wave = tid >> 5;
    const int lane = tid & 31;
    const int lo   = lane & 15;
    const int hi   = lane >> 4;
    const int row0 = blockIdx.y * 128;
    const int col0 = blockIdx.x * 64;

    // Per-thread staging coordinates (16B chunks; 8 chunks per 128B row).
    const int arow = tid >> 3;
    const int aseg = tid & 7;

    // Issue one stage (6 async b128 per thread) into ring buffer `buf`.
    auto issue_stage = [&](int buf, int k0) {
        #pragma unroll
        for (int c = 0; c < 4; ++c) {               // A: 128 rows x 128B
            int row = arow + c * 32;
            u32 loff = lds_offset(&As[buf][row * LDK + aseg * 8]);
            async_copy_b128(loff, A + (size_t)(row0 + row) * K + k0 + aseg * 8);
        }
        #pragma unroll
        for (int c = 0; c < 2; ++c) {               // B: 64 rows x 128B
            int row = arow + c * 32;
            u32 loff = lds_offset(&Bs[buf][row * LDK + aseg * 8]);
            async_copy_b128(loff, Bt + (size_t)(col0 + row) * K + k0 + aseg * 8);
        }
    };

    v8f acc[4];
#pragma unroll
    for (int nt = 0; nt < 4; ++nt)
#pragma unroll
        for (int i = 0; i < 8; ++i) acc[nt][i] = 0.0f;

    // Load all fragments of a stage, then fire 8 back-to-back wmma.
    auto compute_stage = [&](int buf) {
        v16bf afr[2], bfr[8];
        #pragma unroll
        for (int s = 0; s < 2; ++s) {
            const u16* ap = &As[buf][(wave * 16 + lo) * LDK + s * 32 + 8 * hi];
            uint4 a0 = *reinterpret_cast<const uint4*>(ap);
            uint4 a1 = *reinterpret_cast<const uint4*>(ap + 16);
            v8u au = {a0.x, a0.y, a0.z, a0.w, a1.x, a1.y, a1.z, a1.w};
            afr[s] = __builtin_bit_cast(v16bf, au);
        }
        #pragma unroll
        for (int nt = 0; nt < 4; ++nt)
            #pragma unroll
            for (int s = 0; s < 2; ++s) {
                const u16* bp = &Bs[buf][(nt * 16 + lo) * LDK + s * 32 + 8 * hi];
                uint4 b0 = *reinterpret_cast<const uint4*>(bp);
                uint4 b1 = *reinterpret_cast<const uint4*>(bp + 16);
                v8u bu = {b0.x, b0.y, b0.z, b0.w, b1.x, b1.y, b1.z, b1.w};
                bfr[nt * 2 + s] = __builtin_bit_cast(v16bf, bu);
            }
        #pragma unroll
        for (int s = 0; s < 2; ++s)
            #pragma unroll
            for (int nt = 0; nt < 4; ++nt)
                acc[nt] = __builtin_amdgcn_wmma_f32_16x16x32_bf16(
                    false, afr[s], false, bfr[nt * 2 + s], (short)0,
                    acc[nt], false, false);
    };

    const int KT = K / BK;
    issue_stage(0, 0);                              // prologue: tile 0

    // Steady state: branch-free body, one barrier per stage.
    // Writes in stage kt target ring slot (kt+1)%3, last read in stage kt-2;
    // the single barrier of stage kt-1 orders those reads before these writes.
    for (int kt = 0; kt < KT - 1; ++kt) {
        issue_stage((kt + 1) % 3, (kt + 1) * BK);
        WAIT_ASYNCCNT(6);                           // retire tile kt
        __syncthreads();
        compute_stage(kt % 3);
    }
    WAIT_ASYNCCNT(0);                               // last tile
    __syncthreads();
    compute_stage((KT - 1) % 3);

    // ---- epilogue: C layout m = i + 8*hi, n = lo ----
#pragma unroll
    for (int nt = 0; nt < 4; ++nt) {
        int gc = col0 + nt * 16 + lo;
#pragma unroll
        for (int i = 0; i < 8; ++i) {
            int gr = row0 + wave * 16 + 8 * hi + i;
            size_t idx = (size_t)gr * N + gc;
            float v = acc[nt][i];
            if constexpr (EPI == EPI_F32) {
                Cf[idx] = v;
            } else if constexpr (EPI == EPI_ADD) {
                Cf[idx] = v + Res[idx];
            } else if constexpr (EPI == EPI_RELU2B) {
                float t = fmaxf(v, 0.0f);
                Cb[idx] = to_bf16(t * t);
            } else if constexpr (EPI == EPI_SIG) {
                Cf[idx] = 1.0f / (1.0f + expf(-v));
            } else {  // EPI_FINAL: out = x1 + rf * (kf @ Wv)
                Cf[idx] = Res[idx] + Scl[idx] * v;
            }
        }
    }
}

// ---------------------------------------------------------------------------
// Host-side orchestration
// ---------------------------------------------------------------------------
extern "C" void kernel_launch(void* const* d_in, const int* in_sizes, int n_in,
                              void* d_out, int out_size, void* d_ws, size_t ws_size,
                              hipStream_t stream) {
    (void)in_sizes; (void)n_in; (void)out_size; (void)ws_size;

    const float* x     = (const float*)d_in[0];
    const float* lag   = (const float*)d_in[1];
    const float* lab   = (const float*)d_in[2];
    const float* lfg   = (const float*)d_in[3];
    const float* lfb   = (const float*)d_in[4];
    const float* td    = (const float*)d_in[5];
    const float* tf    = (const float*)d_in[6];
    const float* amk   = (const float*)d_in[7];
    const float* amv   = (const float*)d_in[8];
    const float* amr   = (const float*)d_in[9];
    const float* wk    = (const float*)d_in[10];
    const float* wv    = (const float*)d_in[11];
    const float* wr    = (const float*)d_in[12];
    const float* wo    = (const float*)d_in[13];
    const float* fmk   = (const float*)d_in[14];
    const float* fmr   = (const float*)d_in[15];
    const float* fwk   = (const float*)d_in[16];
    const float* fwv   = (const float*)d_in[17];
    const float* fwr   = (const float*)d_in[18];

    char* ws = (char*)d_ws;
    constexpr size_t MB = 1ull << 20;
    float* h     = (float*)(ws +   0 * MB);   //  64 MB f32 (reused both LNs)
    u16*   xk    = (u16*)  (ws +  64 * MB);   //  32 MB bf16
    u16*   xv    = (u16*)  (ws +  96 * MB);   //  32 MB bf16
    u16*   xr    = (u16*)  (ws + 128 * MB);   //  32 MB bf16
    float* kbuf  = (float*)(ws + 160 * MB);   //  64 MB f32
    float* vbuf  = (float*)(ws + 224 * MB);   //  64 MB f32
    float* rbuf  = (float*)(ws + 288 * MB);   //  64 MB f32
    u16*   rwkv  = (u16*)  (ws + 352 * MB);   //  32 MB bf16
    float* x1    = (float*)(ws + 384 * MB);   //  64 MB f32
    u16*   wkT   = (u16*)  (ws + 448 * MB);   //   2 MB
    u16*   wvT   = (u16*)  (ws + 450 * MB);   //   2 MB
    u16*   wrT   = (u16*)  (ws + 452 * MB);   //   2 MB
    u16*   woT   = (u16*)  (ws + 454 * MB);   //   2 MB
    u16*   fwkT  = (u16*)  (ws + 456 * MB);   //   8 MB
    u16*   fwvT  = (u16*)  (ws + 464 * MB);   //   8 MB
    u16*   fwrT  = (u16*)  (ws + 472 * MB);   //   2 MB
    u16*   kf    = (u16*)kbuf;                // 128 MB alias over kbuf+vbuf
    float* rf    = rbuf;                      //  64 MB alias

    // ---- 1. weight convert+transpose to bf16 (K-contiguous B tiles) ----
    {
        int n1 = SIZE_ * ATT_;   // 1M
        int n2 = SIZE_ * LIN_;   // 4M
        transpose_bf16_kernel<<<(n1 + 255) / 256, 256, 0, stream>>>(wk,  wkT,  SIZE_, ATT_);
        transpose_bf16_kernel<<<(n1 + 255) / 256, 256, 0, stream>>>(wv,  wvT,  SIZE_, ATT_);
        transpose_bf16_kernel<<<(n1 + 255) / 256, 256, 0, stream>>>(wr,  wrT,  SIZE_, ATT_);
        transpose_bf16_kernel<<<(n1 + 255) / 256, 256, 0, stream>>>(wo,  woT,  ATT_,  SIZE_);
        transpose_bf16_kernel<<<(n2 + 255) / 256, 256, 0, stream>>>(fwk, fwkT, SIZE_, LIN_);
        transpose_bf16_kernel<<<(n2 + 255) / 256, 256, 0, stream>>>(fwv, fwvT, LIN_,  SIZE_);
        transpose_bf16_kernel<<<(n1 + 255) / 256, 256, 0, stream>>>(fwr, fwrT, SIZE_, SIZE_);
    }

    const int elemBlocks = NTOK * SIZE_ / 256;            // 65536
    const dim3 g1024(ATT_ / 64, NTOK / 128);              // (16, 128)
    const dim3 g4096(LIN_ / 64, NTOK / 128);              // (64, 128)

    // ---- 2. attention branch ----
    ln_kernel<<<NTOK, 256, 0, stream>>>(x, lag, lab, h);
    mix_kernel<true><<<elemBlocks, 256, 0, stream>>>(h, amk, amv, amr, xk, xv, xr);
    gemm_bf16<EPI_F32><<<g1024, 256, 0, stream>>>(xk, wkT, kbuf, nullptr, nullptr, nullptr, NTOK, ATT_, SIZE_);
    gemm_bf16<EPI_F32><<<g1024, 256, 0, stream>>>(xv, wvT, vbuf, nullptr, nullptr, nullptr, NTOK, ATT_, SIZE_);
    gemm_bf16<EPI_F32><<<g1024, 256, 0, stream>>>(xr, wrT, rbuf, nullptr, nullptr, nullptr, NTOK, ATT_, SIZE_);
    wkv_kernel<<<(B_ * ATT_) / 256, 256, 0, stream>>>(kbuf, vbuf, rbuf, td, tf, rwkv);
    gemm_bf16<EPI_ADD><<<g1024, 256, 0, stream>>>(rwkv, woT, x1, nullptr, x, nullptr, NTOK, SIZE_, ATT_);

    // ---- 3. channel-mixing branch ----
    ln_kernel<<<NTOK, 256, 0, stream>>>(x1, lfg, lfb, h);
    mix_kernel<false><<<elemBlocks, 256, 0, stream>>>(h, fmk, nullptr, fmr, xk, nullptr, xr);
    gemm_bf16<EPI_RELU2B><<<g4096, 256, 0, stream>>>(xk, fwkT, nullptr, kf, nullptr, nullptr, NTOK, LIN_, SIZE_);
    gemm_bf16<EPI_SIG><<<g1024, 256, 0, stream>>>(xr, fwrT, rf, nullptr, nullptr, nullptr, NTOK, SIZE_, SIZE_);
    gemm_bf16<EPI_FINAL><<<g1024, 256, 0, stream>>>(kf, fwvT, (float*)d_out, nullptr, x1, rf, NTOK, SIZE_, LIN_);
}
